// LatentEncoder_89962384982449
// MI455X (gfx1250) — matrix operand order repaired and checked by reading, not compiled, not run on hardware
//
#include <hip/hip_runtime.h>
#include <hip/hip_bf16.h>
#include <math.h>

typedef __attribute__((ext_vector_type(16))) __bf16 bf16x16;
typedef __attribute__((ext_vector_type(8)))  __bf16 bf16x8;
typedef __attribute__((ext_vector_type(8)))  float  f32x8;

#define BATCH 4096
#define FEAT  1024
#define NY    64
#define NL    256            // n_logits = 2 * N_Z
#define KSPLIT 4
#define JPER  (NY / KSPLIT)  // 16 experts per block
#define MTILE 128
#define NTILE 128
#define FHALF 512            // hiddens LDS tile covers half of FEAT
#define CHUNKS (FHALF / 32)  // 16 K-chunks resident
#define MAIN_TOT (2 * JPER * CHUNKS)   // 512 K-steps per block

__device__ __forceinline__ void load16(const float* __restrict__ p, float v[16]) {
  const float4* p4 = reinterpret_cast<const float4*>(p);
  float4 a = p4[0], b = p4[1], c = p4[2], d = p4[3];
  v[0]=a.x;  v[1]=a.y;  v[2]=a.z;  v[3]=a.w;
  v[4]=b.x;  v[5]=b.y;  v[6]=b.z;  v[7]=b.w;
  v[8]=c.x;  v[9]=c.y;  v[10]=c.z; v[11]=c.w;
  v[12]=d.x; v[13]=d.y; v[14]=d.z; v[15]=d.w;
}

__global__ __launch_bounds__(256) void gemm_partial_kernel(
    const float* __restrict__ hid, const float* __restrict__ y,
    const float* __restrict__ W,   const float* __restrict__ bias,
    float* __restrict__ part)
{
  // A cache: [chunk(16)][mSub(8)][lane(32)][2 x bf16x8] = 128 KB, WMMA fragment order
  __shared__ bf16x8 hidA[CHUNKS * 8 * 32 * 2];
  // B double buffer: [buf][nSub(8)][lane(32)][2 x bf16x8] = 16 KB
  __shared__ bf16x8 lB[2][8 * 32 * 2];
  // y stage, transposed: [jLocal(16)][row(128)] = 8 KB
  __shared__ float yT[JPER * MTILE];

  const int t    = threadIdx.x;
  const int lane = t & 31;
  const int w    = t >> 5;
  const int mw   = w >> 1;        // 0..3 -> 2 M-subtiles
  const int nw   = w & 1;         // 0..1 -> 4 N-subtiles
  const int mHi  = (lane >> 4) << 3;   // C/D: M = v + 8*lane[4]
  const int nLo  = lane & 15;

  const int rowBase = blockIdx.x * MTILE;
  const int nBase   = blockIdx.y * NTILE;
  const int kz      = blockIdx.z;
  const int jBase   = kz * JPER;

  const int r = t >> 1;           // 0..127: fill row (A) / fill column (B)
  const int h = t & 1;            // K-half16 of a 32-wide chunk
  const int mSubF = r >> 4, mF = r & 15;   // fill-side A coords
  const int nSubF = r >> 4, nF = r & 15;   // fill-side B coords

  f32x8 acc[2][4];
  #pragma unroll
  for (int i = 0; i < 2; ++i)
    #pragma unroll
    for (int jn = 0; jn < 4; ++jn)
      acc[i][jn] = (f32x8)0.0f;

  // ---- stage y[rowBase:+128, jBase:+16] into LDS, transposed to [j][row] ----
  {
    const float4* yp = reinterpret_cast<const float4*>(y + (size_t)(rowBase + r) * NY + jBase + h * 8);
    float4 v0 = yp[0], v1 = yp[1];
    float tmp[8] = {v0.x, v0.y, v0.z, v0.w, v1.x, v1.y, v1.z, v1.w};
    #pragma unroll
    for (int q = 0; q < 8; ++q) yT[(h * 8 + q) * MTILE + r] = tmp[q];
  }

  // ---- pack 16 fp32 -> one 128x32 bf16 A chunk slice (fragment layout) ----
  auto packA = [&](int chunk, const float av[16]) {
    bf16x8 alo, ahi;
    #pragma unroll
    for (int q = 0; q < 8; ++q) { alo[q] = (__bf16)av[q]; ahi[q] = (__bf16)av[q + 8]; }
    hidA[((chunk * 8 + mSubF) * 32 + mF + 16 * h) * 2 + 0] = alo;
    hidA[((chunk * 8 + mSubF) * 32 + mF + 16 * h) * 2 + 1] = ahi;
  };

  auto fillHid = [&](int half) {
    const float* base = hid + (size_t)(rowBase + r) * FEAT + half * FHALF + h * 16;
    #pragma unroll 4
    for (int c = 0; c < CHUNKS; ++c) {
      float av[16];
      load16(base + c * 32, av);
      packA(c, av);
    }
  };

  // ---- B-tile fill (double buffered): it -> (half, jLocal, chunk) ----
  auto fillB = [&](int p, int it) {
    const int j  = jBase + ((it >> 4) & (JPER - 1));
    const int f0 = ((it >> 8) << 9) + ((it & (CHUNKS - 1)) << 5);
    float bv[16];
    load16(W + ((size_t)j * NL + nBase + r) * FEAT + f0 + h * 16, bv);
    bf16x8 b0, b1;
    #pragma unroll
    for (int q = 0; q < 8; ++q) { b0[q] = (__bf16)bv[q]; b1[q] = (__bf16)bv[q + 8]; }
    lB[p][(nSubF * 32 + nF + 16 * h) * 2 + 0] = b0;
    lB[p][(nSubF * 32 + nF + 16 * h) * 2 + 1] = b1;
  };

  // ---- one K=32 step into the per-expert accumulator (init: C = 0) ----
  auto consume = [&](int p, int c, f32x8 (&accj)[2][4], bool init) {
    bf16x16 a0 = *reinterpret_cast<const bf16x16*>(&hidA[((c * 8 + mw * 2 + 0) * 32 + lane) * 2]);
    bf16x16 a1 = *reinterpret_cast<const bf16x16*>(&hidA[((c * 8 + mw * 2 + 1) * 32 + lane) * 2]);
    #pragma unroll
    for (int jn = 0; jn < 4; ++jn) {
      bf16x16 bf = *reinterpret_cast<const bf16x16*>(&lB[p][((nw * 4 + jn) * 32 + lane) * 2]);
      f32x8 c0 = init ? (f32x8)0.0f : accj[0][jn];
      f32x8 c1 = init ? (f32x8)0.0f : accj[1][jn];
      accj[0][jn] = __builtin_amdgcn_wmma_f32_16x16x32_bf16(false, a0, false, bf, (short)0, c0, false, false);
      accj[1][jn] = __builtin_amdgcn_wmma_f32_16x16x32_bf16(false, a1, false, bf, (short)0, c1, false, false);
    }
  };

  // ---- fold one expert segment:  acc += y[i,j] * accj  (fp32, per-row scale) ----
  auto fold = [&](int jl, const f32x8 (&accj)[2][4]) {
    #pragma unroll
    for (int i = 0; i < 2; ++i) {
      const float4* yp = reinterpret_cast<const float4*>(&yT[jl * MTILE + (mw * 2 + i) * 16 + mHi]);
      float4 ya = yp[0], yb = yp[1];
      const float yv[8] = {ya.x, ya.y, ya.z, ya.w, yb.x, yb.y, yb.z, yb.w};
      #pragma unroll
      for (int jn = 0; jn < 4; ++jn)
        #pragma unroll
        for (int v = 0; v < 8; ++v)
          acc[i][jn][v] += yv[v] * accj[i][jn][v];
    }
  };

  // ---- main loop: 2 f-halves x 16 experts x 16 chunks ----
  fillB(0, 0);
  int p = 0, it = 0;
  for (int half = 0; half < 2; ++half) {
    __syncthreads();                 // prior reads of hidA done
    fillHid(half);
    __syncthreads();                 // hidA (and pending lB) visible
    for (int jl = 0; jl < JPER; ++jl) {
      f32x8 accj[2][4];
      // chunk 0 (peeled): start segment with C = 0
      if (it + 1 < MAIN_TOT) fillB(p ^ 1, it + 1);
      consume(p, 0, accj, true);
      __syncthreads();
      p ^= 1; ++it;
      for (int c = 1; c < CHUNKS; ++c, ++it) {
        if (it + 1 < MAIN_TOT) fillB(p ^ 1, it + 1);
        consume(p, c, accj, false);
        __syncthreads();
        p ^= 1;
      }
      fold(jl, accj);
    }
  }

  // ---- bias GEMM  y @ b  (kz==0 only): two K=32 steps, reuse hidA chunk 0 ----
  if (kz == 0) {
    for (int bi = 0; bi < 2; ++bi) {
      __syncthreads();
      { // A tile from y
        float av[16];
        load16(y + (size_t)(rowBase + r) * NY + bi * 32 + h * 16, av);
        packA(0, av);
      }
      { // B tile from b (strided rows)
        float bv[16];
        #pragma unroll
        for (int e = 0; e < 16; ++e)
          bv[e] = bias[(size_t)(bi * 32 + 16 * h + e) * NL + nBase + r];
        bf16x8 b0, b1;
        #pragma unroll
        for (int q = 0; q < 8; ++q) { b0[q] = (__bf16)bv[q]; b1[q] = (__bf16)bv[q + 8]; }
        lB[0][(nSubF * 32 + nF + 16 * h) * 2 + 0] = b0;
        lB[0][(nSubF * 32 + nF + 16 * h) * 2 + 1] = b1;
      }
      __syncthreads();
      bf16x16 a0 = *reinterpret_cast<const bf16x16*>(&hidA[((mw * 2 + 0) * 32 + lane) * 2]);
      bf16x16 a1 = *reinterpret_cast<const bf16x16*>(&hidA[((mw * 2 + 1) * 32 + lane) * 2]);
      #pragma unroll
      for (int jn = 0; jn < 4; ++jn) {
        bf16x16 bf = *reinterpret_cast<const bf16x16*>(&lB[0][((nw * 4 + jn) * 32 + lane) * 2]);
        acc[0][jn] = __builtin_amdgcn_wmma_f32_16x16x32_bf16(false, a0, false, bf, (short)0, acc[0][jn], false, false);
        acc[1][jn] = __builtin_amdgcn_wmma_f32_16x16x32_bf16(false, a1, false, bf, (short)0, acc[1][jn], false, false);
      }
    }
  }

  // ---- write fp32 partial (C/D layout: M = v + 8*lane[4], N = lane[3:0]) ----
  float* out = part + (size_t)kz * BATCH * NL;
  #pragma unroll
  for (int i = 0; i < 2; ++i) {
    const int mSub = mw * 2 + i;
    #pragma unroll
    for (int jn = 0; jn < 4; ++jn) {
      const int col = nBase + (nw * 4 + jn) * 16 + nLo;
      #pragma unroll
      for (int v = 0; v < 8; ++v) {
        const int row = rowBase + mSub * 16 + mHi + v;
        out[(size_t)row * NL + col] = acc[i][jn][v];
      }
    }
  }
}

// Reduce KSPLIT partials (fixed order -> deterministic) + Gaussian head epilogue.
__global__ __launch_bounds__(256) void reduce_act_kernel(
    const float* __restrict__ part, float* __restrict__ out)
{
  const int idx = (blockIdx.x * 256 + threadIdx.x) * 4;   // 0 .. BATCH*NL-1, step 4
  float4 s = *reinterpret_cast<const float4*>(part + idx);
  #pragma unroll
  for (int kz = 1; kz < KSPLIT; ++kz) {
    float4 q = *reinterpret_cast<const float4*>(part + (size_t)kz * BATCH * NL + idx);
    s.x += q.x; s.y += q.y; s.z += q.z; s.w += q.w;
  }
  const int i = idx >> 8;        // / NL
  const int k = idx & (NL - 1);  // 4-aligned; never straddles the mu/sigma split
  if (k < NL / 2) {
    *reinterpret_cast<float4*>(out + (size_t)i * (NL / 2) + k) = s;                 // mu
  } else {
    float4 rr;
    float v[4] = {s.x, s.y, s.z, s.w}, o[4];
    #pragma unroll
    for (int e = 0; e < 4; ++e) {
      const float sp = (v[e] > 20.0f) ? v[e] : log1pf(__expf(v[e]));                // softplus
      o[e] = sqrtf(sp);
    }
    rr.x = o[0]; rr.y = o[1]; rr.z = o[2]; rr.w = o[3];
    *reinterpret_cast<float4*>(out + (size_t)BATCH * (NL / 2) + (size_t)i * (NL / 2) + (k - NL / 2)) = rr;
  }
}

extern "C" void kernel_launch(void* const* d_in, const int* in_sizes, int n_in,
                              void* d_out, int out_size, void* d_ws, size_t ws_size,
                              hipStream_t stream) {
  const float* hiddens = (const float*)d_in[0];   // [4096, 1024]
  const float* y       = (const float*)d_in[1];   // [4096, 64]
  const float* W       = (const float*)d_in[2];   // [64, 256, 1024]
  const float* bias    = (const float*)d_in[3];   // [64, 256]
  float* out  = (float*)d_out;                    // mu[4096,128] ++ sigma[4096,128]
  float* part = (float*)d_ws;                     // KSPLIT * 4096 * 256 fp32 = 16 MB

  dim3 grid(BATCH / MTILE, NL / NTILE, KSPLIT);   // 32 x 2 x 4 = 256 workgroups
  hipLaunchKernelGGL(gemm_partial_kernel, grid, dim3(256), 0, stream,
                     hiddens, y, W, bias, part);
  hipLaunchKernelGGL(reduce_act_kernel, dim3(BATCH * NL / (256 * 4)), dim3(256), 0, stream,
                     part, out);
}
